// MMD_GCN_67156108640417
// MI455X (gfx1250) — compile-verified, matrix operand order checked by reading, not run on hardware
//
#include <hip/hip_runtime.h>
#include <hip/hip_bf16.h>

typedef __attribute__((ext_vector_type(16))) __bf16 v16bf;
typedef __attribute__((ext_vector_type(8)))  float  v8f;

#define LDS_STRIDE 136   // bf16 elements per W row in LDS (128 + pad)

// ---------------------------------------------------------------------------
// GEMM: H[i,o] = sum_k A[i,k] * W[o,k], A = scale[k]*X[i,k]+shift[k] (optional)
// Block = 256 threads = 8 waves; each wave owns a 16-row strip; block = 128 rows.
// W staged to LDS as bf16 once per block. v_wmma_f32_16x16x32_bf16, K-loop of 4,
// 8 column tiles fully unrolled (32 WMMAs straight-line).
// ---------------------------------------------------------------------------
__global__ __launch_bounds__(256) void gcn_gemm_wmma(
    const float* __restrict__ X, const float* __restrict__ W,
    float* __restrict__ H, const float* __restrict__ sc,
    const float* __restrict__ sh, int N)
{
    __shared__ __bf16 Wl[128 * LDS_STRIDE];
    __shared__ float  scs[128], shs[128];

    int tid = threadIdx.x;
    for (int idx = tid; idx < 128 * 128; idx += 256) {
        int o = idx >> 7, k = idx & 127;
        Wl[o * LDS_STRIDE + k] = (__bf16)W[idx];
    }
    if (tid < 128) {
        scs[tid] = sc ? sc[tid] : 1.0f;
        shs[tid] = sh ? sh[tid] : 0.0f;
    }
    __syncthreads();

    int wave     = tid >> 5;
    int lane     = tid & 31;
    int laneRow  = lane & 15;
    int laneHalf = lane >> 4;
    int rowBase  = blockIdx.x * 128 + wave * 16;
    // wave-uniform copy in an SGPR so bounds tests become scalar branches
    int rowBaseU = __builtin_amdgcn_readfirstlane(rowBase);
    int row      = rowBase + laneRow;
    bool rowOK   = row < N;
    bool fullStrip = (rowBaseU + 16 <= N);       // scalar condition
    const float* xrow = X + (size_t)(rowOK ? row : 0) * 128;

    // A operand: 4 K-chunks of 32; per-lane layout per CDNA5 ISA 16-bit A 16x32.
    v16bf a[4];
    #pragma unroll
    for (int c = 0; c < 4; ++c) {
        int kb = c * 32;
        #pragma unroll
        for (int j = 0; j < 16; ++j) {
            int k = kb + ((j < 8) ? 0 : 16) + laneHalf * 8 + (j & 7);
            float v = rowOK ? fmaf(scs[k], xrow[k], shs[k]) : 0.0f;
            a[c][j] = (__bf16)v;
        }
    }

    int mBase = laneHalf * 8;                    // D rows this lane holds
    // lane<16 -> n=laneRow, m=r ; lane>=16 -> n=laneRow, m=8+r
    float* hcol = H + (size_t)(rowBase + mBase) * 128 + laneRow;

    #pragma unroll
    for (int ct = 0; ct < 8; ++ct) {             // 8 column tiles of 16
        int colBase = ct * 16;
        int o = colBase + laneRow;               // B[k,n] = W[colBase+n, k]
        v8f acc = {0.f, 0.f, 0.f, 0.f, 0.f, 0.f, 0.f, 0.f};
        #pragma unroll
        for (int c = 0; c < 4; ++c) {
            int kb = c * 32;
            v16bf bv;
            #pragma unroll
            for (int j = 0; j < 16; ++j) {
                int k = kb + ((j < 8) ? 0 : 16) + laneHalf * 8 + (j & 7);
                bv[j] = Wl[o * LDS_STRIDE + k];
            }
            acc = __builtin_amdgcn_wmma_f32_16x16x32_bf16(
                false, a[c], false, bv, (short)0, acc, false, false);
        }
        if (fullStrip) {                         // scalar branch: no exec masking
            #pragma unroll
            for (int r = 0; r < 8; ++r)
                hcol[(size_t)r * 128 + colBase] = acc[r];
        } else {
            #pragma unroll
            for (int r = 0; r < 8; ++r) {
                int orow = rowBase + mBase + r;
                if (orow < N) H[(size_t)orow * 128 + colBase + laneRow] = acc[r];
            }
        }
    }
}

// ---------------------------------------------------------------------------
// Degree / normalization prep
// ---------------------------------------------------------------------------
__global__ void init_deg(float* deg, int N) {
    int i = blockIdx.x * blockDim.x + threadIdx.x;
    if (i < N) deg[i] = 1.0f;                       // self loop
}

__global__ void accum_deg(const long long* __restrict__ ei, float* deg, int E) {
    int stride = gridDim.x * blockDim.x;
    for (int e = blockIdx.x * blockDim.x + threadIdx.x; e < E; e += stride)
        atomicAdd(&deg[ei[(size_t)E + e]], 1.0f);   // targets = ei[1]
}

__global__ void deg_to_dis(float* deg, int N) {
    int i = blockIdx.x * blockDim.x + threadIdx.x;
    if (i < N) deg[i] = rsqrtf(deg[i]);             // deg >= 1 always
}

// out[i,:] = dis[i]^2 * h[i,:]   (self-loop term; also initializes out)
__global__ void init_out(const float* __restrict__ h, const float* __restrict__ dis,
                         float* __restrict__ out, int N)
{
    long long total = (long long)N * 32;
    long long stride = (long long)gridDim.x * blockDim.x;
    for (long long i = blockIdx.x * (long long)blockDim.x + threadIdx.x; i < total; i += stride) {
        int node = (int)(i >> 5);
        float d = dis[node]; float s = d * d;
        float4 v = ((const float4*)h)[i];
        v.x *= s; v.y *= s; v.z *= s; v.w *= s;
        ((float4*)out)[i] = v;
    }
}

// wave-per-edge gather/scale/scatter-add (L2-resident, atomic f32)
__global__ __launch_bounds__(256) void scatter_edges(
    const long long* __restrict__ ei, const float* __restrict__ h,
    const float* __restrict__ dis, float* __restrict__ out, int E)
{
    int gw   = (blockIdx.x * blockDim.x + threadIdx.x) >> 5;
    int lane = threadIdx.x & 31;
    int nw   = (gridDim.x * blockDim.x) >> 5;
    for (int e = gw; e < E; e += nw) {
        long long r = ei[e];
        long long c = ei[(size_t)E + e];
        float nrm = dis[r] * dis[c];
        float4 v = ((const float4*)(h + (size_t)r * 128))[lane];
        float* ob = out + (size_t)c * 128 + lane * 4;
        atomicAdd(ob + 0, nrm * v.x);
        atomicAdd(ob + 1, nrm * v.y);
        atomicAdd(ob + 2, nrm * v.z);
        atomicAdd(ob + 3, nrm * v.w);
    }
}

__global__ void zero_stats(float* stats) { stats[threadIdx.x] = 0.0f; }

// bias add + per-node mean subtract (in place) + per-feature sum/sumsq accumulate
__global__ __launch_bounds__(256) void center_stats(
    float* __restrict__ out, const float* __restrict__ bias,
    float* __restrict__ stats, int N)
{
    int gw   = (blockIdx.x * blockDim.x + threadIdx.x) >> 5;
    int lane = threadIdx.x & 31;
    int nw   = (gridDim.x * blockDim.x) >> 5;
    float4 bb = ((const float4*)bias)[lane];
    float s0 = 0, s1 = 0, s2 = 0, s3 = 0;
    float q0 = 0, q1 = 0, q2 = 0, q3 = 0;
    for (int i = gw; i < N; i += nw) {
        float4 v = ((float4*)out)[(size_t)i * 32 + lane];
        v.x += bb.x; v.y += bb.y; v.z += bb.z; v.w += bb.w;
        float ps = v.x + v.y + v.z + v.w;
        for (int m = 16; m; m >>= 1) ps += __shfl_xor(ps, m, 32);
        float mean = ps * (1.0f / 128.0f);
        v.x -= mean; v.y -= mean; v.z -= mean; v.w -= mean;
        ((float4*)out)[(size_t)i * 32 + lane] = v;
        s0 += v.x; s1 += v.y; s2 += v.z; s3 += v.w;
        q0 += v.x * v.x; q1 += v.y * v.y; q2 += v.z * v.z; q3 += v.w * v.w;
    }
    int f = lane * 4;
    atomicAdd(&stats[f + 0], s0); atomicAdd(&stats[f + 1], s1);
    atomicAdd(&stats[f + 2], s2); atomicAdd(&stats[f + 3], s3);
    atomicAdd(&stats[128 + f + 0], q0); atomicAdd(&stats[128 + f + 1], q1);
    atomicAdd(&stats[128 + f + 2], q2); atomicAdd(&stats[128 + f + 3], q3);
}

// scale[f] = gamma*rsqrt(var+eps); shift[f] = beta - mu*scale  (consumed by next GEMM)
__global__ void finalize_stats(const float* __restrict__ stats,
                               const float* __restrict__ gamma,
                               const float* __restrict__ beta,
                               float* __restrict__ scsh, float invN)
{
    int f = threadIdx.x;
    float mu  = stats[f] * invN;
    float var = stats[128 + f] * invN - mu * mu;
    float s   = gamma[f] * rsqrtf(var + 1e-5f);
    scsh[f]       = s;
    scsh[128 + f] = beta[f] - mu * s;
}

__global__ void apply_affine(float* __restrict__ out, const float* __restrict__ scsh,
                             long long total)
{
    long long stride = (long long)gridDim.x * blockDim.x;
    for (long long i = blockIdx.x * (long long)blockDim.x + threadIdx.x; i < total; i += stride) {
        int f = (int)(i & 127);
        out[i] = scsh[f] * out[i] + scsh[128 + f];
    }
}

extern "C" void kernel_launch(void* const* d_in, const int* in_sizes, int n_in,
                              void* d_out, int out_size, void* d_ws, size_t ws_size,
                              hipStream_t stream)
{
    const float*     x     = (const float*)d_in[0];
    const long long* ei    = (const long long*)d_in[1];
    const float*     W     = (const float*)d_in[2];
    const float*     b     = (const float*)d_in[3];
    const float*     gamma = (const float*)d_in[4];
    const float*     beta  = (const float*)d_in[5];

    int N = in_sizes[0] / 128;
    int E = in_sizes[1] / 2;
    int L = in_sizes[2] / (128 * 128);

    float* h     = (float*)d_ws;                 // N*128 f32
    float* deg   = h + (size_t)N * 128;          // N f32 (becomes dis)
    float* stats = deg + N;                      // 256 f32 (sum | sumsq)
    float* scsh  = stats + 256;                  // 256 f32 (scale | shift)
    float* out   = (float*)d_out;                // N*128 f32 aggregation buffer

    // normalization coefficients
    init_deg<<<(N + 255) / 256, 256, 0, stream>>>(deg, N);
    accum_deg<<<2048, 256, 0, stream>>>(ei, deg, E);
    deg_to_dis<<<(N + 255) / 256, 256, 0, stream>>>(deg, N);
    const float* dis = deg;

    const float* xin = x;
    const float* sc  = nullptr;
    const float* sh  = nullptr;
    int gemmGrid = (N + 127) / 128;

    for (int l = 0; l < L; ++l) {
        gcn_gemm_wmma<<<gemmGrid, 256, 0, stream>>>(
            xin, W + (size_t)l * 128 * 128, h, sc, sh, N);
        init_out<<<8192, 256, 0, stream>>>(h, dis, out, N);
        scatter_edges<<<4096, 256, 0, stream>>>(ei, h, dis, out, E);
        zero_stats<<<1, 256, 0, stream>>>(stats);
        center_stats<<<1024, 256, 0, stream>>>(out, b + (size_t)l * 128, stats, N);
        finalize_stats<<<1, 128, 0, stream>>>(stats, gamma + (size_t)l * 128,
                                              beta + (size_t)l * 128, scsh,
                                              1.0f / (float)N);
        xin = out; sc = scsh; sh = scsh + 128;
    }
    // final BN affine of last layer, in place on d_out
    apply_affine<<<2048, 256, 0, stream>>>(out, scsh, (long long)N * 128);
}
